// cosine_seqNet_54700703482458
// MI455X (gfx1250) — matrix operand order repaired and checked by reading, not compiled, not run on hardware
//
#include <hip/hip_runtime.h>

// ---------------------------------------------------------------------------
// cosine_seqNet on MI455X (gfx1250, wave32, WMMA, async global->LDS copies)
//
// Math reduction:
//   out[b,o] = conv_b[o] + (1/6) * sum_{i,k} conv_w[o,i,k] * A[b, i*5+k]
//   A[b,i*5+k] = sum_{t=0..5} w[b,t+k]*x[b,t+k,i]
//   w[b,s] = dot(x[b,s],x[b,6]) / (max(|x[b,s]|,eps)*max(|x[b,6]|,eps))
//
// Heavy op: C(256x4096) = A(256x20480) x W^T(20480x4096)  -> bf16 WMMA GEMM
// with double-buffered LDS tiles filled by GLOBAL_LOAD_ASYNC_TO_LDS_B128.
// ---------------------------------------------------------------------------

typedef __bf16 bf16;
typedef __attribute__((ext_vector_type(16))) __bf16 v16bf;
typedef __attribute__((ext_vector_type(4)))  __bf16 v4bf;
typedef __attribute__((ext_vector_type(8)))  float  v8f;

#define BATCH 256
#define SEQ   10
#define DIN   4096
#define DOUT  4096
#define WIN   5
#define KDIM  (DIN * WIN)      // 20480
#define EPS   1e-8f

static __device__ __forceinline__ bf16 to_bf16(float f) {
    unsigned int u = __builtin_bit_cast(unsigned int, f);
    u += 0x7FFFu + ((u >> 16) & 1u);          // round-to-nearest-even
    unsigned short h = (unsigned short)(u >> 16);
    return __builtin_bit_cast(bf16, h);
}

// async DMA-style copy: 16B global -> LDS, tracked by ASYNCcnt (no VGPR hop)
static __device__ __forceinline__ void async_ld_b128(unsigned lds_off, const bf16* g) {
    asm volatile("global_load_async_to_lds_b128 %0, %1, off"
                 :
                 : "v"(lds_off), "v"((unsigned long long)(uintptr_t)g)
                 : "memory");
}

// ---------------------------------------------------------------------------
// Kernel 0: conv_w fp32 -> bf16 (168 MB image, ~fits in 192 MB L2)
// ---------------------------------------------------------------------------
__global__ __launch_bounds__(256) void convert_w_kernel(const float* __restrict__ w,
                                                        bf16* __restrict__ wb,
                                                        int n4) {
    int idx    = blockIdx.x * blockDim.x + threadIdx.x;
    int stride = gridDim.x * blockDim.x;
    const float4* src = (const float4*)w;
    v4bf*         dst = (v4bf*)wb;
    for (int i = idx; i < n4; i += stride) {
        float4 v = src[i];
        v4bf o;
        o[0] = to_bf16(v.x); o[1] = to_bf16(v.y);
        o[2] = to_bf16(v.z); o[3] = to_bf16(v.w);
        dst[i] = o;
    }
}

// ---------------------------------------------------------------------------
// Kernel 1: cosine weights + length-6 sliding-window sums -> A[b, i*5+k] bf16
// ---------------------------------------------------------------------------
__global__ __launch_bounds__(256) void prep_kernel(const float* __restrict__ x,
                                                   bf16* __restrict__ A) {
    const int b   = blockIdx.x;
    const int tid = threadIdx.x;
    const float* xb = x + (size_t)b * SEQ * DIN;

    __shared__ float sdot[SEQ];
    __shared__ float snrm[SEQ];
    __shared__ float sw[SEQ];

    float dot[SEQ], nrm[SEQ];
#pragma unroll
    for (int s = 0; s < SEQ; ++s) { dot[s] = 0.f; nrm[s] = 0.f; }

#pragma unroll 4
    for (int ii = 0; ii < DIN / 256; ++ii) {
        int i = tid + ii * 256;
        float xc = xb[6 * DIN + i];
#pragma unroll
        for (int s = 0; s < SEQ; ++s) {
            float v = xb[s * DIN + i];
            dot[s] += v * xc;
            nrm[s] += v * v;
        }
    }

    if (tid < SEQ) { sdot[tid] = 0.f; snrm[tid] = 0.f; }
    __syncthreads();
#pragma unroll
    for (int s = 0; s < SEQ; ++s) {
        atomicAdd(&sdot[s], dot[s]);   // ds_add_f32
        atomicAdd(&snrm[s], nrm[s]);
    }
    __syncthreads();
    if (tid < SEQ) {
        float ns = sqrtf(snrm[tid]);
        float nc = sqrtf(snrm[6]);
        sw[tid]  = sdot[tid] / (fmaxf(ns, EPS) * fmaxf(nc, EPS));
    }
    __syncthreads();

    bf16* Ab = A + (size_t)b * KDIM;
#pragma unroll 4
    for (int ii = 0; ii < DIN / 256; ++ii) {
        int i = tid + ii * 256;
        float c[SEQ + 1];
        c[0] = 0.f;
#pragma unroll
        for (int s = 0; s < SEQ; ++s)
            c[s + 1] = c[s] + xb[s * DIN + i] * sw[s];
#pragma unroll
        for (int k = 0; k < WIN; ++k)
            Ab[i * WIN + k] = to_bf16(c[k + 6] - c[k]);   // sum of 6 terms
    }
}

// ---------------------------------------------------------------------------
// Kernel 2: C(256x4096) = A(256x20480,bf16) x Wb^T + bias, scaled by 1/6.
// Block tile 128x128, 8 waves as 2(M) x 4(N); wave tile 64x32 = 4x2 WMMAs.
// Double-buffered LDS tiles filled with async global->LDS b128 copies.
// ---------------------------------------------------------------------------
#define TM 128
#define TN 128
#define TK 32
#define TILEB (TM * TK * 2)   // 8192 bytes per tile buffer

__global__ __launch_bounds__(256) void gemm_wmma_kernel(const bf16* __restrict__ Ag,
                                                        const bf16* __restrict__ Wb,
                                                        const float* __restrict__ bias,
                                                        float* __restrict__ out) {
    __shared__ __align__(16) bf16 As[2][TM * TK];   // 2 x 8 KB
    __shared__ __align__(16) bf16 Bs[2][TN * TK];   // 2 x 8 KB

    const int tid  = threadIdx.x;
    const int lane = tid & 31;
    const int wid  = tid >> 5;
    const int wm   = wid & 1;        // 0..1  (M)
    const int wn   = wid >> 1;       // 0..3  (N)
    const int m0   = blockIdx.x * TM;
    const int n0   = blockIdx.y * TN;

    const int mloc = lane & 15;      // row/col within a 16x16 tile
    const int half = lane >> 4;      // lane group 0/1
    const int kb   = half * 8;       // A-frag K base is 2*kb (k = 16*half + s)

    // each thread owns one 32B span (two b128 chunks) of each 8 KB tile
    const int r   = tid >> 1;            // tile row 0..127
    const int cc0 = (tid & 1) * 2;       // first 16B chunk in the 64B row

    // LDS byte offsets (ISA: LDS address = low 32 bits of generic address)
    const unsigned ldsA = (unsigned)(uintptr_t)&As[0][r * TK + cc0 * 8];
    const unsigned ldsB = (unsigned)(uintptr_t)&Bs[0][r * TK + cc0 * 8];
    const bf16* gA = Ag + (m0 + r) * KDIM + cc0 * 8;
    const bf16* gB = Wb + (n0 + r) * KDIM + cc0 * 8;

    v8f acc[4][2] = {};

    // prologue: start DMA of tile 0 into buffer 0
    async_ld_b128(ldsA,      gA);
    async_ld_b128(ldsA + 16, gA + 8);
    async_ld_b128(ldsB,      gB);
    async_ld_b128(ldsB + 16, gB + 8);

    int cur = 0;
    for (int k0 = 0; k0 < KDIM; k0 += TK) {
        if (k0 + TK < KDIM) {
            // kick off next tile into the other buffer, then wait only for
            // the current tile's 4 copies (async ops complete in order)
            const unsigned bo = (unsigned)((cur ^ 1) * TILEB);
            async_ld_b128(ldsA + bo,      gA + k0 + TK);
            async_ld_b128(ldsA + bo + 16, gA + k0 + TK + 8);
            async_ld_b128(ldsB + bo,      gB + k0 + TK);
            async_ld_b128(ldsB + bo + 16, gB + k0 + TK + 8);
            if (k0 + 2 * TK < KDIM)      // keep the weight stream warm in L2
                __builtin_prefetch(gB + k0 + 2 * TK, 0, 1);
            asm volatile("s_wait_asynccnt 0x4" ::: "memory");
        } else {
            asm volatile("s_wait_asynccnt 0x0" ::: "memory");
        }
        __syncthreads();

        // ---- B fragments (K contiguous 16 halves per lane-half) ----
        v16bf bfrag[2];
#pragma unroll
        for (int nt = 0; nt < 2; ++nt)
            bfrag[nt] = *(const v16bf*)(&Bs[cur][(wn * 32 + nt * 16 + mloc) * TK + half * 16]);

        // ---- A fragments + WMMA ----
#pragma unroll
        for (int mt = 0; mt < 4; ++mt) {
            const bf16* rp = &As[cur][(wm * 64 + mt * 16 + mloc) * TK];
            v16bf a;
#pragma unroll
            for (int j = 0; j < 8; ++j) {
                a[j]     = rp[2 * kb + j];        // K = 2kb..2kb+7
                a[j + 8] = rp[2 * kb + 8 + j];    // K = 2kb+8..2kb+15
            }
#pragma unroll
            for (int nt = 0; nt < 2; ++nt)
                acc[mt][nt] = __builtin_amdgcn_wmma_f32_16x16x32_bf16(
                    false, a, false, bfrag[nt], (short)0, acc[mt][nt], false, false);
        }
        __syncthreads();
        cur ^= 1;
    }

    // ---- epilogue: out = bias + acc/6 ; C/D layout M = r + 8*half, N = mloc
    const float inv6 = 1.0f / 6.0f;
#pragma unroll
    for (int mt = 0; mt < 4; ++mt)
#pragma unroll
        for (int nt = 0; nt < 2; ++nt) {
            int col = n0 + wn * 32 + nt * 16 + mloc;
            float bv = bias[col];
#pragma unroll
            for (int rr = 0; rr < 8; ++rr) {
                int row = m0 + wm * 64 + mt * 16 + rr + half * 8;
                out[row * DOUT + col] = bv + acc[mt][nt][rr] * inv6;
            }
        }
}

// ---------------------------------------------------------------------------
extern "C" void kernel_launch(void* const* d_in, const int* in_sizes, int n_in,
                              void* d_out, int out_size, void* d_ws, size_t ws_size,
                              hipStream_t stream) {
    const float* x      = (const float*)d_in[0];   // (256,10,4096)
    const float* conv_w = (const float*)d_in[1];   // (4096,4096,5)
    const float* conv_b = (const float*)d_in[2];   // (4096,)
    float*       out    = (float*)d_out;           // (256,4096)

    const size_t WN = (size_t)DOUT * KDIM;         // 83,886,080 elements
    bf16* Wb = (bf16*)d_ws;                        // 167,772,160 B
    bf16* A  = (bf16*)((char*)d_ws + WN * sizeof(bf16));  // + 10,485,760 B

    convert_w_kernel<<<dim3(4096), dim3(256), 0, stream>>>(conv_w, Wb, (int)(WN / 4));
    prep_kernel<<<dim3(BATCH), dim3(256), 0, stream>>>(x, A);
    gemm_wmma_kernel<<<dim3(BATCH / TM, DOUT / TN), dim3(256), 0, stream>>>(A, Wb, conv_b, out);
}
